// GaitDecoder_60138132078596
// MI455X (gfx1250) — matrix operand order repaired and checked by reading, not compiled, not run on hardware
//
#include <hip/hip_runtime.h>
#include <math.h>

// ---- problem constants (from reference) ----
#define BB   16
#define LATN 256
#define DD   256
#define DI   512
#define DSN  16
#define DTR  16
#define LL   1024
#define KK   4
#define OUTN 64
#define NL   4
#define MM   (BB*LL)   // 16384 rows

typedef _Float16 f16;
typedef _Float16 f16x8 __attribute__((ext_vector_type(8)));
typedef _Float16 v16h  __attribute__((ext_vector_type(16)));
typedef float    v8f   __attribute__((ext_vector_type(8)));
typedef unsigned int u32x4 __attribute__((ext_vector_type(4)));
typedef int          i32x4 __attribute__((ext_vector_type(4)));
typedef int          i32x8 __attribute__((ext_vector_type(8)));

// Tensor Data Mover availability (device pass only; host pass -> fallback path)
#ifdef __HIP_DEVICE_COMPILE__
# if __has_builtin(__builtin_amdgcn_tensor_load_to_lds) && __has_builtin(__builtin_amdgcn_s_wait_tensorcnt)
#   define GD_TDM 1
# endif
#endif
#if __has_include(<hip/amd_detail/amd_gfx1250_TDM.h>)
# define GD_TDM_6ARG 1
#endif

// ---------------- pack f32 -> f16 ----------------
__global__ void gd_pack_f16(const float* __restrict__ src, f16* __restrict__ dst, int n) {
    int i = blockIdx.x * blockDim.x + threadIdx.x;
    if (i < n) dst[i] = (f16)src[i];
}

// ---------------- base = gelu(LN(z) @ latent_w.T + bias) ----------------
__global__ void gd_base(const float* __restrict__ z, const float* __restrict__ lg,
                        const float* __restrict__ lb, const float* __restrict__ lw,
                        const float* __restrict__ lbias, float* __restrict__ base) {
    __shared__ float sx[LATN];
    __shared__ float red[256];
    int b = blockIdx.x, t = threadIdx.x;
    float v = z[b * LATN + t];
    red[t] = v; __syncthreads();
    for (int s = 128; s > 0; s >>= 1) { if (t < s) red[t] += red[t + s]; __syncthreads(); }
    float mu = red[0] * (1.0f / LATN); __syncthreads();
    float d = v - mu;
    red[t] = d * d; __syncthreads();
    for (int s = 128; s > 0; s >>= 1) { if (t < s) red[t] += red[t + s]; __syncthreads(); }
    float var = red[0] * (1.0f / LATN); __syncthreads();
    sx[t] = d * rsqrtf(var + 1e-5f) * lg[t] + lb[t];
    __syncthreads();
    float acc = lbias[t];
    const float* wrow = lw + (size_t)t * LATN;
    for (int k = 0; k < LATN; ++k) acc += sx[k] * wrow[k];
    base[b * DD + t] = 0.5f * acc * (1.0f + erff(acc * 0.70710678f)); // exact gelu
}

// ---------------- h[b,l,d] = base[b,d] + time_emb[l,d] ----------------
__global__ void gd_hinit(const float* __restrict__ base, const float* __restrict__ temb,
                         float* __restrict__ h) {
    int i = blockIdx.x * 256 + threadIdx.x;          // < 16*1024*256
    int d = i & 255, l = (i >> 8) & 1023, b = i >> 18;
    h[i] = base[b * DD + d] + temb[l * DD + d];
}

// ---------------- row LayerNorm (D=256) -> f16 ----------------
__global__ void gd_ln_rows(const float* __restrict__ x, const float* __restrict__ g,
                           const float* __restrict__ b, f16* __restrict__ y) {
    __shared__ float red[256];
    int row = blockIdx.x, t = threadIdx.x;
    float v = x[(size_t)row * DD + t];
    red[t] = v; __syncthreads();
    for (int s = 128; s > 0; s >>= 1) { if (t < s) red[t] += red[t + s]; __syncthreads(); }
    float mu = red[0] * (1.0f / DD); __syncthreads();
    float d = v - mu;
    red[t] = d * d; __syncthreads();
    for (int s = 128; s > 0; s >>= 1) { if (t < s) red[t] += red[t + s]; __syncthreads(); }
    float var = red[0] * (1.0f / DD);
    y[(size_t)row * DD + t] = (f16)(d * rsqrtf(var + 1e-5f) * g[t] + b[t]);
}

// ---------------- WMMA GEMM: C[M,N] = A[M,K] * W[N,K]^T (+bias) ----------------
// Block = 4 waves = 128 rows x 16 cols. The 16xKtot W panel (shared by all 4
// waves) is staged into LDS once per block via the Tensor Data Mover (with the
// TDM pad feature inserting an 8-halfword row pad to spread LDS banks), or a
// cooperative copy fallback. Each wave computes two 16x16 M-tiles so each B
// fragment read from LDS feeds two v_wmma ops.
#define SPAD 8                        // f16 elements of LDS row padding
__global__ __launch_bounds__(128)
void gd_wmma_gemm(const f16* __restrict__ A, const f16* __restrict__ W,
                  const float* __restrict__ bias, float* __restrict__ C,
                  float* __restrict__ Cres, int Ktot, int Ncols) {
    __shared__ __align__(128) f16 sW[16 * (512 + SPAD)];   // max Ktot = 512
    const int SST = Ktot + SPAD;     // padded LDS row stride (f16 elems)
    int lane = threadIdx.x;          // 0..31
    int wv   = threadIdx.y;          // 0..3
    int tile_n = blockIdx.x;
    int hl  = lane >> 4;
    int r16 = lane & 15;

    // ---- stage W tile (16 rows x Ktot) into LDS ----
#ifdef GD_TDM
    if (wv == 0) {
        unsigned long long ga = (unsigned long long)(uintptr_t)(W + (size_t)tile_n * 16 * Ktot);
        unsigned lds_off = (unsigned)(uintptr_t)(void*)sW;   // low 32 bits = LDS byte addr
        unsigned pint = (Ktot == 512) ? 7u : 6u;             // row bytes = 2^(pint+3)
        // D# group0: count=1 | lds_addr | global_addr | type=2
        u32x4 g0 = { 1u, lds_off, (unsigned)(ga & 0xffffffffu),
                     (unsigned)((ga >> 32) & 0x1ffffffu) | 0x80000000u };
        // D# group1: data_size=2B, pad_enable, pad_interval, pad_amount=4 dwords,
        //            tensor_dim0=Ktot, tensor_dim1=16, tile_dim0=Ktot, tile_dim1=16,
        //            tensor_dim0_stride=Ktot
        i32x8 g1 = { (int)((1u << 16) | (1u << 20) | (pint << 22) | (3u << 25)),
                     (int)(((unsigned)Ktot & 0xffffu) << 16),
                     (int)(16u << 16),
                     (int)((unsigned)Ktot << 16),
                     16,
                     Ktot,
                     0, 0 };
        i32x4 gz = { 0, 0, 0, 0 };
#  ifdef GD_TDM_6ARG
        i32x8 gz8 = { 0, 0, 0, 0, 0, 0, 0, 0 };
        __builtin_amdgcn_tensor_load_to_lds(g0, g1, gz, gz, gz8, 0);
#  else
        __builtin_amdgcn_tensor_load_to_lds(g0, g1, gz, gz, 0);
#  endif
        __builtin_amdgcn_s_wait_tensorcnt(0);
    }
    __syncthreads();
#else
    {
        int rowsz = Ktot >> 3;                               // f16x8 chunks per row
        for (int idx = wv * 32 + lane; idx < 16 * rowsz; idx += 128) {
            int rr = idx / rowsz, cc = idx - rr * rowsz;
            *(f16x8*)(sW + rr * SST + cc * 8) =
                *(const f16x8*)(W + ((size_t)tile_n * 16 + rr) * Ktot + cc * 8);
        }
        __syncthreads();
    }
#endif

    // ---- compute: two 16x16 M-tiles per wave ----
    int mbase = (blockIdx.y * 4 + wv) * 32;
    const f16* Ap0 = A + (size_t)(mbase + r16) * Ktot;
    const f16* Ap1 = Ap0 + (size_t)16 * Ktot;
    const f16* Bp  = sW + (size_t)r16 * SST;
    v8f acc0 = {}, acc1 = {};
    for (int kb = 0; kb < Ktot; kb += 32) {
        // A 16x32 f16 layout: elems 0..7 -> K=kb+8h+e ; elems 8..15 -> K=kb+16+8h+e
        f16x8 a00 = *(const f16x8*)(Ap0 + kb + 8 * hl);
        f16x8 a01 = *(const f16x8*)(Ap0 + kb + 16 + 8 * hl);
        f16x8 a10 = *(const f16x8*)(Ap1 + kb + 8 * hl);
        f16x8 a11 = *(const f16x8*)(Ap1 + kb + 16 + 8 * hl);
        // B 32x16 f16 layout: elems 0..7 -> K=kb+8h+e ; elems 8..15 -> K=kb+16+8h+e (from LDS)
        f16x8 b0 = *(const f16x8*)(Bp + kb + 8 * hl);
        f16x8 b1 = *(const f16x8*)(Bp + kb + 16 + 8 * hl);
        __builtin_prefetch(Ap0 + kb + 64, 0, 1);
        __builtin_prefetch(Ap1 + kb + 64, 0, 1);
        v16h av0 = __builtin_shufflevector(a00, a01, 0,1,2,3,4,5,6,7,8,9,10,11,12,13,14,15);
        v16h av1 = __builtin_shufflevector(a10, a11, 0,1,2,3,4,5,6,7,8,9,10,11,12,13,14,15);
        v16h bv  = __builtin_shufflevector(b0,  b1,  0,1,2,3,4,5,6,7,8,9,10,11,12,13,14,15);
        acc0 = __builtin_amdgcn_wmma_f32_16x16x32_f16(false, av0, false, bv,
                                                      (short)0, acc0, false, false);
        acc1 = __builtin_amdgcn_wmma_f32_16x16x32_f16(false, av1, false, bv,
                                                      (short)0, acc1, false, false);
    }
    int n = tile_n * 16 + r16;
    float bb = bias ? bias[n] : 0.0f;
    for (int r = 0; r < 8; ++r) {                 // D layout: VGPR r -> row 8*h + r
        int m0 = mbase + hl * 8 + r;
        int m1 = m0 + 16;
        size_t i0 = (size_t)m0 * Ncols + n;
        size_t i1 = (size_t)m1 * Ncols + n;
        float v0 = acc0[r] + bb, v1 = acc1[r] + bb;
        if (Cres) { Cres[i0] += v0; Cres[i1] += v1; }
        else      { C[i0] = v0;     C[i1] = v1;     }
    }
}

// ---------------- depthwise causal conv (K=4) + SiLU ----------------
__global__ void gd_conv_silu(const float* __restrict__ xz, const float* __restrict__ cw,
                             const float* __restrict__ cb, float* __restrict__ xs,
                             f16* __restrict__ xsbf) {
    int i = blockIdx.x * 256 + threadIdx.x;          // < 16*1024*512
    int c = i & 511, l = (i >> 9) & 1023, b = i >> 19;
    float acc = cb[c];
    for (int k = 0; k < KK; ++k) {
        int lt = l - (KK - 1) + k;
        if (lt >= 0) acc += xz[((size_t)(b * LL + lt)) * (2 * DI) + c] * cw[c * KK + k];
    }
    float s = acc / (1.0f + expf(-acc));
    size_t o = (size_t)(b * LL + l) * DI + c;
    xs[o] = s; xsbf[o] = (f16)s;
}

// ---------------- dt = softplus(dbc[:,:16] @ dt_w.T + dt_b) ----------------
__global__ void gd_dt(const float* __restrict__ dbc, const float* __restrict__ dtw,
                      const float* __restrict__ dtb, float* __restrict__ dt) {
    int i = blockIdx.x * 256 + threadIdx.x;          // < 16384*512
    int di = i & 511, m = i >> 9;
    float acc = dtb[di];
    const float* row = dbc + (size_t)m * (DTR + 2 * DSN);
    for (int r = 0; r < DTR; ++r) acc += row[r] * dtw[di * DTR + r];
    dt[i] = (acc > 20.0f) ? acc : log1pf(expf(acc));
}

// ---------------- selective scan: state (16) in registers per (b,di) ----------------
__global__ void gd_scan(const float* __restrict__ xs, const float* __restrict__ dt,
                        const float* __restrict__ dbc, const float* __restrict__ xz,
                        const float* __restrict__ A_log, const float* __restrict__ Dp,
                        f16* __restrict__ ybf) {
    int gid = blockIdx.x * 256 + threadIdx.x;        // < 16*512
    int di = gid & 511, b = gid >> 9;
    float Ar[DSN], hst[DSN];
    for (int s = 0; s < DSN; ++s) { Ar[s] = -expf(A_log[di * DSN + s]); hst[s] = 0.0f; }
    float Dv = Dp[di];
    for (int t = 0; t < LL; ++t) {
        size_t m = (size_t)b * LL + t;
        float dtv = dt[m * DI + di];
        float xv  = xs[m * DI + di];
        const float* Bt = dbc + m * (DTR + 2 * DSN) + DTR;
        const float* Ct = Bt + DSN;
        float y = 0.0f, dx = dtv * xv;
        for (int s = 0; s < DSN; ++s) {
            hst[s] = __expf(dtv * Ar[s]) * hst[s] + dx * Bt[s];
            y += hst[s] * Ct[s];
        }
        y += xv * Dv;
        float zg = xz[m * (2 * DI) + DI + di];
        ybf[m * DI + di] = (f16)(y * (zg / (1.0f + expf(-zg))));
    }
}

// ============================ host launcher ============================
extern "C" void kernel_launch(void* const* d_in, const int* in_sizes, int n_in,
                              void* d_out, int out_size, void* d_ws, size_t ws_size,
                              hipStream_t stream) {
    const float* z        = (const float*)d_in[0];
    const float* latent_g = (const float*)d_in[1];
    const float* latent_b = (const float*)d_in[2];
    const float* latent_w = (const float*)d_in[3];
    const float* latent_bi= (const float*)d_in[4];
    const float* time_emb = (const float*)d_in[5];
    const float* ln_g     = (const float*)d_in[6];
    const float* ln_b     = (const float*)d_in[7];
    const float* in_w     = (const float*)d_in[8];
    const float* in_b     = (const float*)d_in[9];
    const float* conv_w   = (const float*)d_in[10];
    const float* conv_b   = (const float*)d_in[11];
    const float* xp_w     = (const float*)d_in[12];
    const float* dt_w     = (const float*)d_in[13];
    const float* dt_b     = (const float*)d_in[14];
    const float* A_log    = (const float*)d_in[15];
    const float* Dp       = (const float*)d_in[16];
    const float* out_w    = (const float*)d_in[17];
    const float* out_b    = (const float*)d_in[18];
    const float* on_g     = (const float*)d_in[19];
    const float* on_b     = (const float*)d_in[20];
    const float* op_w     = (const float*)d_in[21];
    const float* op_b     = (const float*)d_in[22];
    float* out = (float*)d_out;

    // workspace arena (256B aligned slots)
    char* ws = (char*)d_ws;
    size_t off = 0;
    auto take = [&](size_t bytes) { size_t o = off; off = (off + bytes + 255) & ~(size_t)255; return o; };
    float* base  = (float*)(ws + take((size_t)BB * DD * 4));
    float* hbuf  = (float*)(ws + take((size_t)MM * DD * 4));
    float* xzbuf = (float*)(ws + take((size_t)MM * 2 * DI * 4));
    float* xsbuf = (float*)(ws + take((size_t)MM * DI * 4));
    f16*   xsbf  = (f16*)  (ws + take((size_t)MM * DI * 2));
    float* dbc   = (float*)(ws + take((size_t)MM * (DTR + 2 * DSN) * 4));
    float* dtbuf = (float*)(ws + take((size_t)MM * DI * 4));
    f16*   ybf   = (f16*)  (ws + take((size_t)MM * DI * 2));
    f16*   lnbf  = (f16*)  (ws + take((size_t)MM * DD * 2));
    f16*   inw16 = (f16*)  (ws + take((size_t)NL * 2 * DI * DD * 2));
    f16*   xpw16 = (f16*)  (ws + take((size_t)NL * (DTR + 2 * DSN) * DI * 2));
    f16*   outw16= (f16*)  (ws + take((size_t)NL * DD * DI * 2));
    f16*   opw16 = (f16*)  (ws + take((size_t)OUTN * DD * 2));
    (void)ws_size; (void)in_sizes; (void)n_in; (void)out_size;

    // 1) pack weights to f16
    { int n = NL * 2 * DI * DD;            gd_pack_f16<<<(n + 255) / 256, 256, 0, stream>>>(in_w,  inw16,  n); }
    { int n = NL * (DTR + 2 * DSN) * DI;   gd_pack_f16<<<(n + 255) / 256, 256, 0, stream>>>(xp_w,  xpw16,  n); }
    { int n = NL * DD * DI;                gd_pack_f16<<<(n + 255) / 256, 256, 0, stream>>>(out_w, outw16, n); }
    { int n = OUTN * DD;                   gd_pack_f16<<<(n + 255) / 256, 256, 0, stream>>>(op_w,  opw16,  n); }

    // 2) base + h init
    gd_base<<<BB, 256, 0, stream>>>(z, latent_g, latent_b, latent_w, latent_bi, base);
    gd_hinit<<<(MM * DD) / 256, 256, 0, stream>>>(base, time_emb, hbuf);

    // 3) mamba layers
    dim3 gblk(32, 4);
    for (int i = 0; i < NL; ++i) {
        gd_ln_rows<<<MM, 256, 0, stream>>>(hbuf, ln_g + i * DD, ln_b + i * DD, lnbf);
        // in_proj: M=16384, K=256, N=1024
        gd_wmma_gemm<<<dim3(2 * DI / 16, MM / 128), gblk, 0, stream>>>(
            lnbf, inw16 + (size_t)i * 2 * DI * DD, in_b + i * 2 * DI,
            xzbuf, nullptr, DD, 2 * DI);
        gd_conv_silu<<<(MM * DI) / 256, 256, 0, stream>>>(
            xzbuf, conv_w + (size_t)i * DI * KK, conv_b + i * DI, xsbuf, xsbf);
        // xp_proj: M=16384, K=512, N=48
        gd_wmma_gemm<<<dim3((DTR + 2 * DSN) / 16, MM / 128), gblk, 0, stream>>>(
            xsbf, xpw16 + (size_t)i * (DTR + 2 * DSN) * DI, nullptr,
            dbc, nullptr, DI, DTR + 2 * DSN);
        gd_dt<<<(MM * DI) / 256, 256, 0, stream>>>(
            dbc, dt_w + (size_t)i * DI * DTR, dt_b + i * DI, dtbuf);
        gd_scan<<<(BB * DI) / 256, 256, 0, stream>>>(
            xsbuf, dtbuf, dbc, xzbuf, A_log + (size_t)i * DI * DSN, Dp + i * DI, ybf);
        // out_proj with fused residual: h += y @ out_w.T + out_b
        gd_wmma_gemm<<<dim3(DD / 16, MM / 128), gblk, 0, stream>>>(
            ybf, outw16 + (size_t)i * DD * DI, out_b + i * DD,
            nullptr, hbuf, DI, DD);
    }

    // 4) final LN + output projection (M=16384, K=256, N=64) -> d_out
    gd_ln_rows<<<MM, 256, 0, stream>>>(hbuf, on_g, on_b, lnbf);
    gd_wmma_gemm<<<dim3(OUTN / 16, MM / 128), gblk, 0, stream>>>(
        lnbf, opw16, op_b, out, nullptr, DD, OUTN);
}
